// InvBlock_77163382440330
// MI455X (gfx1250) — matrix operand order repaired and checked by reading, not compiled, not run on hardware
//
#include <hip/hip_runtime.h>

// ---------------------------------------------------------------------------
// InvBlock (additive coupling with tanh) on MI455X / gfx1250.
// Four chained GEMMs in bf16 WMMA (v_wmma_f32_16x16x32_bf16), f32 accumulate;
// tanh / step updates fused into GEMM epilogues. Wave tile 64x64 (16 acc
// tiles) -> 1.0 b128-load per WMMA; operands indexed off kernel args so the
// compiler infers global address space (global_load_b128, LOADcnt-only
// waits); fast tanh via v_exp_f32/v_rcp_f32; global_prefetch_b8 on streams.
// ---------------------------------------------------------------------------

typedef __bf16 bf16_t;
typedef bf16_t v16bf __attribute__((ext_vector_type(16)));
typedef bf16_t v8bf  __attribute__((ext_vector_type(8)));
typedef float  v8f   __attribute__((ext_vector_type(8)));

#define STEPF 0.1f
#define MM   16384   // batch rows
#define HH   2048    // h
#define HH2  4096    // 2h

__device__ __forceinline__ v8f wmma_bf16(v16bf a, v16bf b, v8f c) {
  return __builtin_amdgcn_wmma_f32_16x16x32_bf16(
      /*neg_a=*/false, a, /*neg_b=*/false, b,
      /*c_mod=*/(short)0, c, /*reuse_a=*/false, /*reuse_b=*/false);
}

// tanh(x) = sign(x) * (1 - e^{-2|x|}) / (1 + e^{-2|x|}); exp2/rcp are TRANS
// ops that co-execute with the matrix pipe.
__device__ __forceinline__ float fast_tanh(float x) {
  const float ax = __builtin_fabsf(x);
  const float t  = __builtin_amdgcn_exp2f(ax * -2.885390081777927f); // -2*log2(e)
  const float r  = (1.0f - t) * __builtin_amdgcn_rcpf(1.0f + t);
  return __builtin_copysignf(r, x);
}

// ---------------------------------------------------------------------------
// Conversion kernels (pure function of d_in; re-run every call)
// ---------------------------------------------------------------------------

// x[:, 0:HH] (row stride HH2) -> bf16 (row stride HH). One block per row.
__global__ __launch_bounds__(256) void cvt_u_kernel(const float* __restrict__ x,
                                                    bf16_t* __restrict__ u) {
  const int row = blockIdx.x;
  const int c   = threadIdx.x * 8;
  const float* src = x + (size_t)row * HH2 + c;
  bf16_t*      dst = u + (size_t)row * HH  + c;
#pragma unroll
  for (int i = 0; i < 8; ++i) dst[i] = (bf16_t)src[i];
}

// W (HH2 x HH f32) -> Wb (bf16, same layout) and Wt (HH x HH2 bf16 transpose)
// so both GEMM directions get a K-contiguous B operand.
__global__ __launch_bounds__(256) void cvt_w_kernel(const float* __restrict__ W,
                                                    bf16_t* __restrict__ Wb,
                                                    bf16_t* __restrict__ Wt) {
  const int n = blockIdx.x;        // 0..HH2-1
  const int k = threadIdx.x * 8;   // 256*8 = HH
#pragma unroll
  for (int i = 0; i < 8; ++i) {
    const float w = W[(size_t)n * HH + k + i];
    Wb[(size_t)n * HH + k + i]    = (bf16_t)w;
    Wt[(size_t)(k + i) * HH2 + n] = (bf16_t)w;
  }
}

// ---------------------------------------------------------------------------
// Tiled WMMA GEMM: D = A(MxK) * Bt^T(KxN) with fused epilogue.
//   MODE 0: out_bf[m][n] = bf16( tanh(acc + bias[n]) )
//   MODE 1: r = oldv[m*old_ld + n] + scale*acc;
//           out_f[m*outf_ld + n] = r;  optionally out_bf[m*N+n] = bf16(r)
// Block: 256 threads = 8 waves (2 M x 4 N); block tile 128(M) x 256(N);
// wave tile 64x64 = 4x4 accumulator tiles. Grid: (N/256, M/128).
//
// Fragment layouts used here (wave32; A and B share contiguous K halves, not the ISA table order):
//  A 16x32: lane<16 -> row m0+lane, K 0..15; lane>=16 -> row
//           m0+lane-16, K 16..31  (two 16B loads)
//  B 32x16: lane<16 -> col n0+lane, K 0..15; lane>=16 -> col n0+lane-16,
//           K 16..31  (one 32B load from the K-contiguous Bt copy)
//  C/D 16x16: VGPR v: lane<16 -> (M=v, N=lane); lane>=16 -> (M=8+v, N=lane-16)
// ---------------------------------------------------------------------------
template <int MODE>
__global__ __launch_bounds__(256) void gemm_wmma_kernel(
    const bf16_t* __restrict__ A,    // M x K row-major
    const bf16_t* __restrict__ Bt,   // N x K row-major (K-contiguous)
    int K, int N,
    const float* __restrict__ bias,  // MODE 0
    const float* __restrict__ oldv,  // MODE 1
    int old_ld, float scale,
    bf16_t* __restrict__ out_bf,     // MODE 0 required; MODE 1 optional
    float* __restrict__ out_f,       // MODE 1
    int outf_ld) {
  const int lane  = threadIdx.x & 31;
  const int wave  = threadIdx.x >> 5;
  const int waveM = wave >> 2;                  // 0..1
  const int waveN = wave & 3;                   // 0..3
  const int wm0 = blockIdx.y * 128 + waveM * 64;
  const int wn0 = blockIdx.x * 256 + waveN * 64;

  v8f acc[4][4] = {};   // [iM][jN]

  // Loop-invariant per-lane element offsets (indexing stays rooted at the
  // __restrict__ kernel args so loads lower to global_load_b128).
  const int lrow = lane & 15;
  const int kha  = (lane >> 4) * 16;   // A K-half offset
  const int khb  = (lane >> 4) * 16;   // B K-half offset
  size_t aoff[4], boff[4];
#pragma unroll
  for (int i = 0; i < 4; ++i)
    aoff[i] = (size_t)(wm0 + i * 16 + lrow) * K + kha;
#pragma unroll
  for (int j = 0; j < 4; ++j)
    boff[j] = (size_t)(wn0 + j * 16 + lrow) * K + khb;

#pragma unroll 2
  for (int k0 = 0; k0 < K; k0 += 32) {
    v16bf a[4], b[4];
#pragma unroll
    for (int j = 0; j < 4; ++j)
      b[j] = *(const v16bf*)(Bt + boff[j] + k0);
#pragma unroll
    for (int i = 0; i < 4; ++i) {
      union { v16bf v; v8bf h[2]; } u;
      u.h[0] = *(const v8bf*)(A + aoff[i] + k0);
      u.h[1] = *(const v8bf*)(A + aoff[i] + k0 + 8);
      a[i] = u.v;
    }
#pragma unroll
    for (int j = 0; j < 4; ++j)
#pragma unroll
      for (int i = 0; i < 4; ++i)
        acc[i][j] = wmma_bf16(a[i], b[j], acc[i][j]);

    // Warm caches ~4KB ahead on representative streams (global_prefetch_b8).
    __builtin_prefetch(A + aoff[0] + k0 + 2048, 0, 1);
    __builtin_prefetch(Bt + boff[0] + k0 + 2048, 0, 1);
  }

  // Epilogue.
  const int ncol    = lane & 15;
  const int rowhalf = (lane >> 4) * 8;
#pragma unroll
  for (int i = 0; i < 4; ++i) {
#pragma unroll
    for (int j = 0; j < 4; ++j) {
      const int n = wn0 + j * 16 + ncol;
#pragma unroll
      for (int v = 0; v < 8; ++v) {
        const int m = wm0 + i * 16 + rowhalf + v;
        const float a = acc[i][j][v];
        if (MODE == 0) {
          const float t = fast_tanh(a + bias[n]);
          out_bf[(size_t)m * N + n] = (bf16_t)t;
        } else {
          const float r = oldv[(size_t)m * old_ld + n] + scale * a;
          out_f[(size_t)m * outf_ld + n] = r;
          if (out_bf) out_bf[(size_t)m * N + n] = (bf16_t)r;
        }
      }
    }
  }
}

// ---------------------------------------------------------------------------
// Launcher
// ---------------------------------------------------------------------------
extern "C" void kernel_launch(void* const* d_in, const int* in_sizes, int n_in,
                              void* d_out, int out_size, void* d_ws, size_t ws_size,
                              hipStream_t stream) {
  const float* x  = (const float*)d_in[0];
  const float* W0 = (const float*)d_in[1];
  const float* b0 = (const float*)d_in[2];
  const float* W1 = (const float*)d_in[3];
  const float* b1 = (const float*)d_in[4];
  float* out = (float*)d_out;

  // Workspace carve-up (all offsets MB-aligned -> 32B alignment holds).
  char* ws = (char*)d_ws;
  bf16_t* u_bf    = (bf16_t*)ws; ws += (size_t)MM  * HH  * sizeof(bf16_t); // 64 MB
  bf16_t* W0_bf   = (bf16_t*)ws; ws += (size_t)HH2 * HH  * sizeof(bf16_t); // 16 MB
  bf16_t* W0T_bf  = (bf16_t*)ws; ws += (size_t)HH  * HH2 * sizeof(bf16_t); // 16 MB
  bf16_t* W1_bf   = (bf16_t*)ws; ws += (size_t)HH2 * HH  * sizeof(bf16_t); // 16 MB
  bf16_t* W1T_bf  = (bf16_t*)ws; ws += (size_t)HH  * HH2 * sizeof(bf16_t); // 16 MB
  bf16_t* mid_bf  = (bf16_t*)ws; ws += (size_t)MM  * HH2 * sizeof(bf16_t); // 128 MB
  bf16_t* vnew_bf = (bf16_t*)ws; ws += (size_t)MM  * HH  * sizeof(bf16_t); // 64 MB

  // ---- conversions ----
  cvt_u_kernel<<<MM, 256, 0, stream>>>(x, u_bf);
  cvt_w_kernel<<<HH2, 256, 0, stream>>>(W0, W0_bf, W0T_bf);
  cvt_w_kernel<<<HH2, 256, 0, stream>>>(W1, W1_bf, W1T_bf);

  const dim3 blk(256);
  const dim3 grid_wide(HH2 / 256, MM / 128);  // N = 4096
  const dim3 grid_half(HH  / 256, MM / 128);  // N = 2048

  // v_mid = tanh(u_old @ W1.T + b1)             (M x 4096, bf16)
  gemm_wmma_kernel<0><<<grid_wide, blk, 0, stream>>>(
      u_bf, W1_bf, /*K=*/HH, /*N=*/HH2, b1,
      nullptr, 0, 0.f, mid_bf, nullptr, 0);

  // v_new = v_old + STEP * (v_mid @ W1)         (M x 2048; f32 -> out[:,h:], bf16 copy)
  gemm_wmma_kernel<1><<<grid_half, blk, 0, stream>>>(
      mid_bf, W1T_bf, /*K=*/HH2, /*N=*/HH, nullptr,
      /*oldv=*/x + HH, /*old_ld=*/HH2, /*scale=*/STEPF,
      vnew_bf, /*out_f=*/out + HH, /*outf_ld=*/HH2);

  // u_mid = tanh(v_new @ W0.T + b0)             (M x 4096, bf16)
  gemm_wmma_kernel<0><<<grid_wide, blk, 0, stream>>>(
      vnew_bf, W0_bf, /*K=*/HH, /*N=*/HH2, b0,
      nullptr, 0, 0.f, mid_bf, nullptr, 0);

  // u_new = u_old - STEP * (u_mid @ W0)         (M x 2048; f32 -> out[:,:h])
  gemm_wmma_kernel<1><<<grid_half, blk, 0, stream>>>(
      mid_bf, W0T_bf, /*K=*/HH2, /*N=*/HH, nullptr,
      /*oldv=*/x, /*old_ld=*/HH2, /*scale=*/-STEPF,
      /*out_bf=*/nullptr, /*out_f=*/out, /*outf_ld=*/HH2);
}
